// HierarchicalLoRALinear_44427141710222
// MI455X (gfx1250) — compile-verified, matrix operand order checked by reading, not run on hardware
//
#include <hip/hip_runtime.h>
#include <hip/hip_bf16.h>

#define IN_DIM   4096
#define OUT_DIM  4096
#define RANK_    16
#define NSUB_    4
#define SCALE_   2.0f          // ALPHA/RANK = 32/16
#define KPAD     96            // 16 + 64 LoRA cols, padded to 3*32
#define BM       128
#define BN       256
#define BK       32
#define LDSS     40            // 32 halfs + 8 pad -> 80B row stride, conflict-free
#define NCHUNK   (IN_DIM / BK + KPAD / BK)   // 128 + 3 = 131

typedef __bf16 bf16_t;
typedef __attribute__((ext_vector_type(16))) __bf16 v16bf;
typedef __attribute__((ext_vector_type(8)))  __bf16 v8bf;
typedef __attribute__((ext_vector_type(4)))  __bf16 v4bf;
typedef __attribute__((ext_vector_type(8)))  float  v8f;
typedef __attribute__((ext_vector_type(4)))  float  v4f;

// ---------------- fp32 -> bf16 bulk convert (x and W) ----------------
__global__ __launch_bounds__(256) void cvt_bf16_kernel(const float* __restrict__ in,
                                                       bf16_t* __restrict__ out,
                                                       size_t n4) {
    size_t i = (size_t)blockIdx.x * blockDim.x + threadIdx.x;
    if (i >= n4) return;
    v4f v = *(const v4f*)(in + i * 4);
    v4bf o;
    o[0] = (bf16_t)v[0]; o[1] = (bf16_t)v[1];
    o[2] = (bf16_t)v[2]; o[3] = (bf16_t)v[3];
    *(v4bf*)(out + i * 4) = o;
}

// ---------------- build Ball[OUT, 96] bf16 (base_B | sub_B | zeros) ----------------
__global__ __launch_bounds__(256) void build_ball_kernel(const float* __restrict__ baseB, // [OUT,16]
                                                         const float* __restrict__ subB,  // [4,OUT,16]
                                                         bf16_t* __restrict__ Ball) {
    int idx = blockIdx.x * blockDim.x + threadIdx.x;
    if (idx >= OUT_DIM * KPAD) return;
    int o = idx / KPAD, c = idx % KPAD;
    float v = 0.f;
    if (c < RANK_) {
        v = baseB[o * RANK_ + c];
    } else if (c < RANK_ + NSUB_ * RANK_) {
        int j = (c - RANK_) >> 4, r = (c - RANK_) & 15;
        v = subB[((size_t)j * OUT_DIM + o) * RANK_ + r];
    }
    Ball[idx] = (bf16_t)v;
}

// ---------------- mid[M, 96] bf16: scaled base-rank + relu-gated sub-ranks ----------------
__global__ __launch_bounds__(256) void mid_kernel(const float* __restrict__ x,     // [M,IN]
                                                  const float* __restrict__ baseA, // [16,IN]
                                                  const float* __restrict__ wgate, // [4,IN]
                                                  const float* __restrict__ subA,  // [64,IN]
                                                  bf16_t* __restrict__ Mid) {
    __shared__ float xs[IN_DIM];
    __shared__ float dots[84];
    const int token = blockIdx.x;
    const float* xr = x + (size_t)token * IN_DIM;
    for (int i = threadIdx.x; i < IN_DIM / 4; i += 256)
        *(v4f*)(xs + i * 4) = *(const v4f*)(xr + i * 4);
    __syncthreads();

    const int wave = threadIdx.x >> 5, lane = threadIdx.x & 31;
    for (int v = wave; v < 84; v += 8) {
        const float* a;
        if (v < 16)       a = baseA + (size_t)v * IN_DIM;
        else if (v < 20)  a = wgate + (size_t)(v - 16) * IN_DIM;
        else              a = subA  + (size_t)(v - 20) * IN_DIM;
        float s = 0.f;
        for (int k = lane * 4; k < IN_DIM; k += 128) {
            v4f xv = *(const v4f*)(xs + k);
            v4f av = *(const v4f*)(a + k);
            s += xv[0] * av[0] + xv[1] * av[1] + xv[2] * av[2] + xv[3] * av[3];
        }
        #pragma unroll
        for (int off = 16; off; off >>= 1) s += __shfl_xor(s, off, 32);
        if (lane == 0) dots[v] = s;
    }
    __syncthreads();

    if (threadIdx.x < KPAD) {
        int c = threadIdx.x;
        float val = 0.f;
        if (c < RANK_) {
            val = SCALE_ * dots[c];
        } else if (c < RANK_ + NSUB_ * RANK_) {
            int j = (c - RANK_) >> 4;
            float g = dots[16 + j];
            g = g > 0.f ? g : 0.f;
            val = SCALE_ * g * dots[20 + (c - RANK_)];
        }
        Mid[(size_t)token * KPAD + c] = (bf16_t)val;
    }
}

// ---------------- main fused GEMM: out = Xb@Wb^T + Mid@Ball^T + bias ----------------
// Block tile 128x256, 8 waves (2x4), wave tile 64x64 (4x4 WMMA accumulators).
// Double-buffered LDS, register-staged global loads, one barrier per K-chunk.
__global__ __launch_bounds__(256) void lora_gemm_kernel(const bf16_t* __restrict__ Xb,   // [M,4096]
                                                        const bf16_t* __restrict__ Wb,   // [OUT,4096]
                                                        const bf16_t* __restrict__ Mid,  // [M,96]
                                                        const bf16_t* __restrict__ Ball, // [OUT,96]
                                                        const float*  __restrict__ bias, // [OUT]
                                                        float* __restrict__ Out) {
    __shared__ bf16_t As[2][BM * LDSS];   // 2 x 5120 halfs
    __shared__ bf16_t Bs[2][BN * LDSS];   // 2 x 10240 halfs

    const int tid  = threadIdx.x;
    const int wave = tid >> 5, lane = tid & 31;
    const int wm = wave & 1, wn = wave >> 1;          // 2 x 4 wave grid
    const int blockMr = blockIdx.y * BM;
    const int blockNr = blockIdx.x * BN;

    v8f acc[4][4] = {};
    v8bf sA[2], sB[4];                                // global->reg staging

    // per-thread chunk coordinates (fixed across chunks)
    const int rA0 = tid >> 2,            kcT = (tid & 3) << 3;   // A: 2 chunks (rows tid>>2, 64+tid>>2)
    const int rA1 = (tid + 256) >> 2;
    const int rB0 = tid >> 2, rB1 = (tid + 256) >> 2;
    const int rB2 = (tid + 512) >> 2, rB3 = (tid + 768) >> 2;

    auto gload = [&](int c) {
        const bf16_t* Asrc;
        const bf16_t* Bsrc;
        int stride, k0;
        if (c < IN_DIM / BK) { Asrc = Xb;  Bsrc = Wb;   stride = IN_DIM; k0 = c * BK; }
        else                 { Asrc = Mid; Bsrc = Ball; stride = KPAD;   k0 = (c - IN_DIM / BK) * BK; }
        sA[0] = *(const v8bf*)(Asrc + (size_t)(blockMr + rA0) * stride + k0 + kcT);
        sA[1] = *(const v8bf*)(Asrc + (size_t)(blockMr + rA1) * stride + k0 + kcT);
        sB[0] = *(const v8bf*)(Bsrc + (size_t)(blockNr + rB0) * stride + k0 + kcT);
        sB[1] = *(const v8bf*)(Bsrc + (size_t)(blockNr + rB1) * stride + k0 + kcT);
        sB[2] = *(const v8bf*)(Bsrc + (size_t)(blockNr + rB2) * stride + k0 + kcT);
        sB[3] = *(const v8bf*)(Bsrc + (size_t)(blockNr + rB3) * stride + k0 + kcT);
    };

    auto lstore = [&](int buf) {
        *(v8bf*)(&As[buf][rA0 * LDSS + kcT]) = sA[0];
        *(v8bf*)(&As[buf][rA1 * LDSS + kcT]) = sA[1];
        *(v8bf*)(&Bs[buf][rB0 * LDSS + kcT]) = sB[0];
        *(v8bf*)(&Bs[buf][rB1 * LDSS + kcT]) = sB[1];
        *(v8bf*)(&Bs[buf][rB2 * LDSS + kcT]) = sB[2];
        *(v8bf*)(&Bs[buf][rB3 * LDSS + kcT]) = sB[3];
    };

    auto load_frag = [&](const bf16_t* smem, int row) -> v16bf {
        const bf16_t* p = smem + row * LDSS + ((lane & 16) ? 8 : 0);
        v8bf lo = *(const v8bf*)(p);        // K 0..7  (or 8..15  for hi half-wave)
        v8bf hi = *(const v8bf*)(p + 16);   // K 16..23 (or 24..31)
        v16bf f;
        #pragma unroll
        for (int e = 0; e < 8; ++e) { f[e] = lo[e]; f[8 + e] = hi[e]; }
        return f;
    };

    auto mma = [&](int buf) {
        v16bf a[4], b[4];
        #pragma unroll
        for (int i = 0; i < 4; ++i)
            a[i] = load_frag(&As[buf][0], wm * 64 + i * 16 + (lane & 15));
        #pragma unroll
        for (int j = 0; j < 4; ++j)
            b[j] = load_frag(&Bs[buf][0], wn * 64 + j * 16 + (lane & 15));
        #pragma unroll
        for (int i = 0; i < 4; ++i)
            #pragma unroll
            for (int j = 0; j < 4; ++j)
                acc[i][j] = __builtin_amdgcn_wmma_f32_16x16x32_bf16(
                    false, a[i], false, b[j], (short)0, acc[i][j], false, false);
    };

    // prologue: stage + store chunk 0
    gload(0);
    lstore(0);
    int cur = 0;
    for (int c = 0; c < NCHUNK; ++c) {
        const bool more = (c + 1) < NCHUNK;
        if (more) gload(c + 1);           // global latency hidden under mma below
        __syncthreads();                  // buf[cur] fully stored; prior reads drained
        mma(cur);
        if (more) lstore(cur ^ 1);
        cur ^= 1;
    }

    // epilogue: C/D layout — VGPR r: lanes0-15 M=r, lanes16-31 M=r+8; N = lane&15
    float bv[4];
    int col[4];
    #pragma unroll
    for (int j = 0; j < 4; ++j) {
        col[j] = blockNr + wn * 64 + j * 16 + (lane & 15);
        bv[j] = bias[col[j]];
    }
    #pragma unroll
    for (int i = 0; i < 4; ++i) {
        int rbase = blockMr + wm * 64 + i * 16 + ((lane >> 4) << 3);
        #pragma unroll
        for (int j = 0; j < 4; ++j) {
            #pragma unroll
            for (int r = 0; r < 8; ++r)
                Out[(size_t)(rbase + r) * OUT_DIM + col[j]] = acc[i][j][r] + bv[j];
        }
    }
}

extern "C" void kernel_launch(void* const* d_in, const int* in_sizes, int n_in,
                              void* d_out, int out_size, void* d_ws, size_t ws_size,
                              hipStream_t stream) {
    const float* x      = (const float*)d_in[0];  // [B,S,IN]
    const float* baseW  = (const float*)d_in[1];  // [OUT,IN]
    const float* baseb  = (const float*)d_in[2];  // [OUT]
    const float* baseA  = (const float*)d_in[3];  // [16,IN]
    const float* baseB  = (const float*)d_in[4];  // [OUT,16]
    const float* wgate  = (const float*)d_in[5];  // [4,IN]
    const float* subA   = (const float*)d_in[6];  // [4,16,IN]
    const float* subB   = (const float*)d_in[7];  // [4,OUT,16]
    float* out = (float*)d_out;

    const int M = in_sizes[0] / IN_DIM;           // 8192 tokens

    // workspace carve-up
    char* ws = (char*)d_ws;
    bf16_t* Xb   = (bf16_t*)ws;                 ws += (size_t)M * IN_DIM * sizeof(bf16_t);
    bf16_t* Wb   = (bf16_t*)ws;                 ws += (size_t)OUT_DIM * IN_DIM * sizeof(bf16_t);
    bf16_t* Mid  = (bf16_t*)ws;                 ws += (size_t)M * KPAD * sizeof(bf16_t);
    bf16_t* Ball = (bf16_t*)ws;

    // 1) convert x and W to bf16
    {
        size_t n4 = (size_t)M * IN_DIM / 4;
        cvt_bf16_kernel<<<(unsigned)((n4 + 255) / 256), 256, 0, stream>>>(x, Xb, n4);
    }
    {
        size_t n4 = (size_t)OUT_DIM * IN_DIM / 4;
        cvt_bf16_kernel<<<(unsigned)((n4 + 255) / 256), 256, 0, stream>>>(baseW, Wb, n4);
    }
    // 2) build folded LoRA B matrix [OUT, 96]
    build_ball_kernel<<<(OUT_DIM * KPAD + 255) / 256, 256, 0, stream>>>(baseB, subB, Ball);
    // 3) per-token mid vector [M, 96] (base rank + relu-gated sub ranks, scaled)
    mid_kernel<<<M, 256, 0, stream>>>(x, baseA, wgate, subA, Mid);
    // 4) fused GEMM with bias (one K-sweep: 4096 dense + 96 folded-LoRA columns)
    dim3 grid(OUT_DIM / BN, M / BM);
    lora_gemm_kernel<<<grid, 256, 0, stream>>>(Xb, Wb, Mid, Ball, baseb, out);
}
